// EnhancedSparseAttention_8813272891551
// MI455X (gfx1250) — matrix operand order repaired and checked by reading, not compiled, not run on hardware
//
#include <hip/hip_runtime.h>
#include <hip/hip_bf16.h>

// ---------------------------------------------------------------------------
// EnhancedSparseAttention for MI455X (gfx1250, wave32, WMMA)
//   B=2, C=512, H=W=48 (L=2304), heads=8, head_dim=64, MAX_REL=32
// Flash-attention design: scores (340MB fp32) are never materialized.
// All matmuls via v_wmma_f32_16x16x32_f16 with fp32 accumulation.
// K/Q tiles stream into LDS via GLOBAL_LOAD_ASYNC_TO_LDS (ASYNCcnt),
// mask gathers are prefetched (global_prefetch_b8) under the WMMA work.
// ---------------------------------------------------------------------------

#define BATCH   2
#define CCH     512
#define HDIM    48
#define LPIX    (HDIM * HDIM)     // 2304
#define NHEADS  8
#define HD      64                // head dim
#define MAXREL  32
#define LN_EPS  1e-5f

typedef __attribute__((ext_vector_type(16))) _Float16 v16h;
typedef __attribute__((ext_vector_type(8)))  float    v8f;

// workspace layout (bytes, all 16B aligned)
#define OFF_WH    ((size_t)0)                               // 4*512*512 f16
#define OFF_XPH   ((size_t)2097152)                         // B*L*C f16
#define OFF_QH    ((size_t)(OFF_XPH + 4718592))
#define OFF_KH    ((size_t)(OFF_QH  + 4718592))
#define OFF_VH    ((size_t)(OFF_KH  + 4718592))
#define OFF_AOH   ((size_t)(OFF_VH  + 4718592))
#define OFF_Y     ((size_t)(OFF_AOH + 4718592))             // B*L*C fp32

// ---- gfx1250 async global->LDS copy (no VGPR staging, tracked by ASYNCcnt) -
__device__ __forceinline__ void async_copy_b64(unsigned lds_off,
                                               unsigned long long gaddr)
{
    asm volatile("global_load_async_to_lds_b64 %0, %1, off"
                 :: "v"(lds_off), "v"(gaddr) : "memory");
}
__device__ __forceinline__ void wait_async0()
{
    asm volatile("s_wait_asynccnt 0x0" ::: "memory");
}

// ------------------------------- prep kernels ------------------------------

__global__ __launch_bounds__(256) void cvt_w_kernel(
    const float* __restrict__ Wq, const float* __restrict__ Wk,
    const float* __restrict__ Wv, const float* __restrict__ Wo,
    _Float16* __restrict__ Wh)
{
    int i = blockIdx.x * 256 + threadIdx.x;           // 4 * 512*512 total
    if (i >= 4 * CCH * CCH) return;
    int m = i >> 18;                                  // which matrix
    int r = i & (CCH * CCH - 1);
    float v;
    if (m == 0)      v = Wq[r];
    else if (m == 1) v = Wk[r];
    else if (m == 2) v = Wv[r];
    else             v = Wo[r];
    Wh[i] = (_Float16)v;
}

// xp = x + pos; write transposed to [B][L][C] f16 (pixel-major for GEMM B tiles)
__global__ __launch_bounds__(256) void prep_x_kernel(
    const float* __restrict__ x, const float* __restrict__ pos,
    _Float16* __restrict__ xph)
{
    size_t i = (size_t)blockIdx.x * 256 + threadIdx.x;  // over B*C*L, l fastest
    if (i >= (size_t)BATCH * CCH * LPIX) return;
    int l = (int)(i % LPIX);
    size_t bc = i / LPIX;
    int c = (int)(bc % CCH);
    int b = (int)(bc / CCH);
    float v = x[i] + pos[(size_t)c * LPIX + l];
    xph[((size_t)b * LPIX + l) * CCH + c] = (_Float16)v;
}

// ------------------------------- QKV GEMM ----------------------------------
// out[o,l] = sum_c W[o,c]*xp[c,l] + bias[o];  M=512, N=2304, K=512 per batch.
// Block: 256 thr (8 waves), 64x64 tile; wave = 16(M) x 32(N) -> 2 WMMA/K-step.

__global__ __launch_bounds__(256) void gemm_qkv_kernel(
    const _Float16* __restrict__ Wh,   // [3][C][C] f16 (q,k,v)
    const _Float16* __restrict__ xph,  // [B][L][C] f16
    const float* __restrict__ bq, const float* __restrict__ bk,
    const float* __restrict__ bv,
    _Float16* __restrict__ qh, _Float16* __restrict__ kh,
    _Float16* __restrict__ vh)         // each [B][L][C] f16
{
    const int which = blockIdx.z % 3;
    const int b     = blockIdx.z / 3;
    const _Float16* W = Wh + (size_t)which * CCH * CCH;
    const float* bias = (which == 0) ? bq : (which == 1) ? bk : bv;
    _Float16* outp    = (which == 0) ? qh : (which == 1) ? kh : vh;

    const int m0 = blockIdx.y * 64;    // channel tile
    const int n0 = blockIdx.x * 64;    // pixel tile
    const int t = threadIdx.x;
    const int wave = t >> 5, lane = t & 31;
    const int mw = wave >> 1, nw = wave & 1;
    const int hi = lane >> 4, lo = lane & 15;

    __shared__ _Float16 As[64 * 36];   // stride 36 f16: conflict-free frag reads
    __shared__ _Float16 Bs[64 * 36];

    v8f acc0 = {}; v8f acc1 = {};
    const _Float16* xb = xph + (size_t)b * LPIX * CCH;

    const int row  = t >> 2;           // 0..63
    const int part = t & 3;            // 8 f16 each

    for (int k0 = 0; k0 < CCH; k0 += 32) {
        // cooperative tile loads (16B global, 8B LDS)
        uint4 va = *reinterpret_cast<const uint4*>(W + (size_t)(m0 + row) * CCH + k0 + part * 8);
        uint4 vb = *reinterpret_cast<const uint4*>(xb + (size_t)(n0 + row) * CCH + k0 + part * 8);
        {
            uint2* pa = reinterpret_cast<uint2*>(&As[row * 36 + part * 8]);
            pa[0] = make_uint2(va.x, va.y); pa[1] = make_uint2(va.z, va.w);
            uint2* pb = reinterpret_cast<uint2*>(&Bs[row * 36 + part * 8]);
            pb[0] = make_uint2(vb.x, vb.y); pb[1] = make_uint2(vb.z, vb.w);
        }
        __syncthreads();

        union { v16h v; unsigned u[8]; } af, bf0, bf1;
        const unsigned* Asu = reinterpret_cast<const unsigned*>(As);
        const unsigned* Bsu = reinterpret_cast<const unsigned*>(Bs);
        const int arow = mw * 16 + lo;
#pragma unroll
        for (int j = 0; j < 8; ++j) {          // A frag: K = 16*hi + 2*j
            int Kb = 16 * hi + 2 * j;
            af.u[j] = Asu[arow * 18 + (Kb >> 1)];
        }
        const int bc0 = nw * 32 + lo;
#pragma unroll
        for (int j = 0; j < 8; ++j) {          // B frag: K = 16*hi + 2*j
            int Kb = 16 * hi + 2 * j;
            bf0.u[j] = Bsu[bc0 * 18 + (Kb >> 1)];
            bf1.u[j] = Bsu[(bc0 + 16) * 18 + (Kb >> 1)];
        }
        acc0 = __builtin_amdgcn_wmma_f32_16x16x32_f16(false, af.v, false, bf0.v,
                                                      (short)0, acc0, false, false);
        acc1 = __builtin_amdgcn_wmma_f32_16x16x32_f16(false, af.v, false, bf1.v,
                                                      (short)0, acc1, false, false);
        __syncthreads();
    }

    // C/D layout: M = r + 8*hi, N = lo
#pragma unroll
    for (int r = 0; r < 8; ++r) {
        int ch = m0 + mw * 16 + r + 8 * hi;
        float bb = bias[ch];
        int l0 = n0 + nw * 32 + lo;
        outp[((size_t)b * LPIX + l0) * CCH + ch]        = (_Float16)(acc0[r] + bb);
        outp[((size_t)b * LPIX + l0 + 16) * CCH + ch]   = (_Float16)(acc1[r] + bb);
    }
}

// --------------------------- flash attention -------------------------------
// Block: 128 thr (4 waves). Each block: 64 query rows of one (b,h).
// Each wave owns a 16-row query strip; loops over 36 key blocks of 64.
// Q/K tiles arrive via async global->LDS; V is transposed through VGPRs.

__global__ __launch_bounds__(128) void attn_kernel(
    const _Float16* __restrict__ qh, const _Float16* __restrict__ kh,
    const _Float16* __restrict__ vh,
    const unsigned char* __restrict__ mask,   // (1,heads,L,L) bool
    const float* __restrict__ rel_table,      // (heads, 65)
    _Float16* __restrict__ aoh)               // [B][L][C] f16
{
    const int b = blockIdx.z, h = blockIdx.y;
    const int q0 = blockIdx.x * 64;
    const int t = threadIdx.x;
    const int wave = t >> 5, lane = t & 31;
    const int hi = lane >> 4, lo = lane & 15;

    __shared__ _Float16 Qs[64 * 68];      // [qrow][d]
    __shared__ _Float16 Ks[64 * 68];      // [key][d]
    __shared__ _Float16 Vs[64 * 68];      // [d][key]  (transposed)
    __shared__ _Float16 Ps[4][16 * 68];   // per-wave P (C-layout -> A-layout hop)

    const int lrow = t >> 1, lhalf = t & 1;   // tile-load roles

    // async-load Q tile (completion folded into first iteration's wait)
    {
        const _Float16* g = qh + ((size_t)(b * LPIX + q0 + lrow)) * CCH + h * HD + lhalf * 32;
        unsigned ldst = (unsigned)(unsigned long long)&Qs[lrow * 68 + lhalf * 32];
#pragma unroll
        for (int j = 0; j < 8; ++j)                    // 8 x b64 = 64B per thread
            async_copy_b64(ldst + j * 8, (unsigned long long)(g + j * 4));
    }

    float m_run[8], l_run[8];
    v8f accd[4] = {};
#pragma unroll
    for (int r = 0; r < 8; ++r) { m_run[r] = -INFINITY; l_run[r] = 0.f; }

    const float scale = 0.125f;   // 1/sqrt(64)
    const unsigned* Qsu = reinterpret_cast<const unsigned*>(Qs);
    const unsigned* Ksu = reinterpret_cast<const unsigned*>(Ks);
    const unsigned* Vsu = reinterpret_cast<const unsigned*>(Vs);
    const unsigned* Psu = reinterpret_cast<const unsigned*>(&Ps[wave][0]);

    for (int kb = 0; kb < LPIX / 64; ++kb) {
        const int m0 = kb * 64;
        __syncthreads();   // prev-iteration consumers of Ks/Vs done
        // async-load K tile [key][d] straight into LDS
        {
            const _Float16* g = kh + ((size_t)(b * LPIX + m0 + lrow)) * CCH + h * HD + lhalf * 32;
            unsigned ldst = (unsigned)(unsigned long long)&Ks[lrow * 68 + lhalf * 32];
#pragma unroll
            for (int j = 0; j < 8; ++j)
                async_copy_b64(ldst + j * 8, (unsigned long long)(g + j * 4));
        }
        // load V tile transposed -> [d][key] (must go through VGPRs)
        {
            union { uint4 q[4]; _Float16 hx[32]; } tmp;
            const uint4* g = reinterpret_cast<const uint4*>(
                vh + ((size_t)(b * LPIX + m0 + lrow)) * CCH + h * HD + lhalf * 32);
#pragma unroll
            for (int j = 0; j < 4; ++j) tmp.q[j] = g[j];
#pragma unroll
            for (int j = 0; j < 32; ++j)
                Vs[(lhalf * 32 + j) * 68 + lrow] = tmp.hx[j];
        }
        wait_async0();
        __syncthreads();

        // prefetch next key block's mask rows while WMMAs run
        if (kb + 1 < LPIX / 64) {
            const unsigned char* mp =
                mask + ((size_t)h * LPIX + q0 + wave * 16 + lo) * LPIX + m0 + 64;
            __builtin_prefetch(mp, 0, 1);
        }

        // S = Q Kt for this wave's 16-row strip: 4 col-tiles x (K=64 -> 2 WMMA)
        float s[4][8];
#pragma unroll
        for (int nt = 0; nt < 4; ++nt) {
            v8f sa = {};
#pragma unroll
            for (int kk = 0; kk < 64; kk += 32) {
                union { v16h v; unsigned u[8]; } af, bf;
#pragma unroll
                for (int j = 0; j < 8; ++j) {
                    int Kb = kk + 16 * hi + 2 * j;
                    af.u[j] = Qsu[(wave * 16 + lo) * 34 + (Kb >> 1)];
                }
#pragma unroll
                for (int j = 0; j < 8; ++j) {
                    int Kb = kk + 16 * hi + 2 * j;
                    bf.u[j] = Ksu[(nt * 16 + lo) * 34 + (Kb >> 1)];
                }
                sa = __builtin_amdgcn_wmma_f32_16x16x32_f16(false, af.v, false, bf.v,
                                                            (short)0, sa, false, false);
            }
            // bias + mask
#pragma unroll
            for (int r = 0; r < 8; ++r) {
                int lg = q0 + wave * 16 + r + 8 * hi;   // global query idx
                int mg = m0 + nt * 16 + lo;             // global key idx
                int d = lg - mg; d = (d > MAXREL) ? MAXREL : (d < -MAXREL ? -MAXREL : d);
                float sv = sa[r] * scale + rel_table[h * (2 * MAXREL + 1) + d + MAXREL];
                if (!mask[((size_t)h * LPIX + lg) * LPIX + mg]) sv = -1e9f;
                s[nt][r] = sv;
            }
        }

        // online softmax (rows live on 16-lane halves -> width-16 shuffles)
#pragma unroll
        for (int r = 0; r < 8; ++r) {
            float mx = fmaxf(fmaxf(s[0][r], s[1][r]), fmaxf(s[2][r], s[3][r]));
#pragma unroll
            for (int off = 8; off >= 1; off >>= 1)
                mx = fmaxf(mx, __shfl_xor(mx, off, 16));
            float mnew = fmaxf(m_run[r], mx);
            float corr = __expf(m_run[r] - mnew);
            m_run[r] = mnew;
            float rs = 0.f;
#pragma unroll
            for (int nt = 0; nt < 4; ++nt) {
                float p = __expf(s[nt][r] - mnew);
                s[nt][r] = p; rs += p;
            }
#pragma unroll
            for (int off = 8; off >= 1; off >>= 1)
                rs += __shfl_xor(rs, off, 16);
            l_run[r] = l_run[r] * corr + rs;
#pragma unroll
            for (int dt = 0; dt < 4; ++dt) accd[dt][r] *= corr;
        }

        // P: C-layout -> LDS -> A-layout
#pragma unroll
        for (int nt = 0; nt < 4; ++nt)
#pragma unroll
            for (int r = 0; r < 8; ++r)
                Ps[wave][(r + 8 * hi) * 68 + nt * 16 + lo] = (_Float16)s[nt][r];
        __syncthreads();   // order P store vs A-frag reads (and Vs reuse)

        // O += P V : 4 d-tiles x (K=64 -> 2 WMMA)
#pragma unroll
        for (int dt = 0; dt < 4; ++dt) {
#pragma unroll
            for (int kk = 0; kk < 64; kk += 32) {
                union { v16h v; unsigned u[8]; } af, bf;
#pragma unroll
                for (int j = 0; j < 8; ++j) {
                    int Kb = kk + 16 * hi + 2 * j;
                    af.u[j] = Psu[lo * 34 + (Kb >> 1)];
                }
#pragma unroll
                for (int j = 0; j < 8; ++j) {
                    int Kb = kk + 16 * hi + 2 * j;
                    bf.u[j] = Vsu[(dt * 16 + lo) * 34 + (Kb >> 1)];
                }
                accd[dt] = __builtin_amdgcn_wmma_f32_16x16x32_f16(false, af.v, false, bf.v,
                                                                  (short)0, accd[dt], false, false);
            }
        }
    }

    // normalize + store (head output into [B][L][C] f16)
#pragma unroll
    for (int dt = 0; dt < 4; ++dt)
#pragma unroll
        for (int r = 0; r < 8; ++r) {
            int lg = q0 + wave * 16 + r + 8 * hi;
            float ov = accd[dt][r] / l_run[r];
            aoh[((size_t)(b * LPIX + lg)) * CCH + h * HD + dt * 16 + lo] = (_Float16)ov;
        }
}

// ------------------------- output proj + residual --------------------------

__global__ __launch_bounds__(256) void gemm_out_kernel(
    const _Float16* __restrict__ Woh,  // [C][C] f16
    const _Float16* __restrict__ aoh,  // [B][L][C] f16
    const float* __restrict__ bo,
    const float* __restrict__ x,       // residual (B,C,H,W)
    float* __restrict__ y)             // [B][L][C] fp32
{
    const int b  = blockIdx.z;
    const int m0 = blockIdx.y * 64;
    const int n0 = blockIdx.x * 64;
    const int t = threadIdx.x;
    const int wave = t >> 5, lane = t & 31;
    const int mw = wave >> 1, nw = wave & 1;
    const int hi = lane >> 4, lo = lane & 15;

    __shared__ _Float16 As[64 * 36];
    __shared__ _Float16 Bs[64 * 36];

    v8f acc0 = {}; v8f acc1 = {};
    const _Float16* ab = aoh + (size_t)b * LPIX * CCH;
    const int row = t >> 2, part = t & 3;

    for (int k0 = 0; k0 < CCH; k0 += 32) {
        uint4 va = *reinterpret_cast<const uint4*>(Woh + (size_t)(m0 + row) * CCH + k0 + part * 8);
        uint4 vb = *reinterpret_cast<const uint4*>(ab + (size_t)(n0 + row) * CCH + k0 + part * 8);
        {
            uint2* pa = reinterpret_cast<uint2*>(&As[row * 36 + part * 8]);
            pa[0] = make_uint2(va.x, va.y); pa[1] = make_uint2(va.z, va.w);
            uint2* pb = reinterpret_cast<uint2*>(&Bs[row * 36 + part * 8]);
            pb[0] = make_uint2(vb.x, vb.y); pb[1] = make_uint2(vb.z, vb.w);
        }
        __syncthreads();
        union { v16h v; unsigned u[8]; } af, bf0, bf1;
        const unsigned* Asu = reinterpret_cast<const unsigned*>(As);
        const unsigned* Bsu = reinterpret_cast<const unsigned*>(Bs);
        const int arow = mw * 16 + lo;
#pragma unroll
        for (int j = 0; j < 8; ++j) {
            int Kb = 16 * hi + 2 * j;
            af.u[j] = Asu[arow * 18 + (Kb >> 1)];
        }
        const int bc0 = nw * 32 + lo;
#pragma unroll
        for (int j = 0; j < 8; ++j) {
            int Kb = 16 * hi + 2 * j;
            bf0.u[j] = Bsu[bc0 * 18 + (Kb >> 1)];
            bf1.u[j] = Bsu[(bc0 + 16) * 18 + (Kb >> 1)];
        }
        acc0 = __builtin_amdgcn_wmma_f32_16x16x32_f16(false, af.v, false, bf0.v,
                                                      (short)0, acc0, false, false);
        acc1 = __builtin_amdgcn_wmma_f32_16x16x32_f16(false, af.v, false, bf1.v,
                                                      (short)0, acc1, false, false);
        __syncthreads();
    }

#pragma unroll
    for (int r = 0; r < 8; ++r) {
        int ch = m0 + mw * 16 + r + 8 * hi;
        float bb = bo[ch];
        int l0 = n0 + nw * 32 + lo;
        float r0 = x[((size_t)b * CCH + ch) * LPIX + l0];
        float r1 = x[((size_t)b * CCH + ch) * LPIX + l0 + 16];
        y[((size_t)b * LPIX + l0) * CCH + ch]      = acc0[r] + bb + r0;
        y[((size_t)b * LPIX + l0 + 16) * CCH + ch] = acc1[r] + bb + r1;
    }
}

// ------------------------------- LayerNorm ---------------------------------
// one wave per pixel; coalesced float4 loads of 512 channels, write (B,C,H,W)

__global__ __launch_bounds__(256) void ln_kernel(
    const float* __restrict__ y, const float* __restrict__ gamma,
    const float* __restrict__ beta, float* __restrict__ out)
{
    const int wave = threadIdx.x >> 5, lane = threadIdx.x & 31;
    const int pix = blockIdx.x * 8 + wave;          // 0 .. B*L-1
    const int b = pix / LPIX, l = pix % LPIX;
    const float4* yp = reinterpret_cast<const float4*>(y + (size_t)pix * CCH);

    float4 vals[4];
    float s = 0.f, sq = 0.f;
#pragma unroll
    for (int k = 0; k < 4; ++k) {
        float4 v = yp[lane + 32 * k];
        vals[k] = v;
        s  += v.x + v.y + v.z + v.w;
        sq += v.x * v.x + v.y * v.y + v.z * v.z + v.w * v.w;
    }
#pragma unroll
    for (int off = 16; off >= 1; off >>= 1) {
        s  += __shfl_xor(s,  off, 32);
        sq += __shfl_xor(sq, off, 32);
    }
    float mean = s * (1.f / CCH);
    float var  = sq * (1.f / CCH) - mean * mean;
    float inv  = rsqrtf(var + LN_EPS);

#pragma unroll
    for (int k = 0; k < 4; ++k) {
        int c0 = (lane + 32 * k) * 4;
        float vv[4] = { vals[k].x, vals[k].y, vals[k].z, vals[k].w };
#pragma unroll
        for (int j = 0; j < 4; ++j) {
            int c = c0 + j;
            out[((size_t)b * CCH + c) * LPIX + l] =
                (vv[j] - mean) * inv * gamma[c] + beta[c];
        }
    }
}

// ------------------------------- launcher ----------------------------------

extern "C" void kernel_launch(void* const* d_in, const int* in_sizes, int n_in,
                              void* d_out, int out_size, void* d_ws, size_t ws_size,
                              hipStream_t stream)
{
    const float* x    = (const float*)d_in[0];
    const unsigned char* mask = (const unsigned char*)d_in[1];   // bool array
    const float* pos  = (const float*)d_in[2];
    const float* Wq   = (const float*)d_in[3];
    const float* bq   = (const float*)d_in[4];
    const float* Wk   = (const float*)d_in[5];
    const float* bk   = (const float*)d_in[6];
    const float* Wv   = (const float*)d_in[7];
    const float* bv   = (const float*)d_in[8];
    const float* Wo   = (const float*)d_in[9];
    const float* bo   = (const float*)d_in[10];
    const float* rel  = (const float*)d_in[11];
    const float* lng  = (const float*)d_in[12];
    const float* lnb  = (const float*)d_in[13];
    float* out        = (float*)d_out;

    char* ws = (char*)d_ws;
    _Float16* Wh  = (_Float16*)(ws + OFF_WH);
    _Float16* xph = (_Float16*)(ws + OFF_XPH);
    _Float16* qh  = (_Float16*)(ws + OFF_QH);
    _Float16* kh  = (_Float16*)(ws + OFF_KH);
    _Float16* vh  = (_Float16*)(ws + OFF_VH);
    _Float16* aoh = (_Float16*)(ws + OFF_AOH);
    float*    yb  = (float*)(ws + OFF_Y);

    cvt_w_kernel<<<(4 * CCH * CCH) / 256, 256, 0, stream>>>(Wq, Wk, Wv, Wo, Wh);
    prep_x_kernel<<<(BATCH * CCH * LPIX) / 256, 256, 0, stream>>>(x, pos, xph);

    dim3 gq(LPIX / 64, CCH / 64, 3 * BATCH);
    gemm_qkv_kernel<<<gq, 256, 0, stream>>>(Wh, xph, bq, bk, bv, qh, kh, vh);

    dim3 ga(LPIX / 64, NHEADS, BATCH);
    attn_kernel<<<ga, 128, 0, stream>>>(qh, kh, vh, mask, rel, aoh);

    dim3 go(LPIX / 64, CCH / 64, BATCH);
    gemm_out_kernel<<<go, 256, 0, stream>>>(Wh + (size_t)3 * CCH * CCH, aoh, bo, x, yb);

    ln_kernel<<<(BATCH * LPIX) / 8, 256, 0, stream>>>(yb, lng, lnb, out);
}
